// Phase2BKCoreModel_36507222016700
// MI455X (gfx1250) — matrix operand order repaired and checked by reading, not compiled, not run on hardware
//
#include <hip/hip_runtime.h>
#include <hip/hip_bf16.h>
#include <math.h>

typedef _Float16 f16;
typedef __attribute__((ext_vector_type(16))) _Float16 v16h;
typedef __attribute__((ext_vector_type(8)))  float    v8f;

#define D_MODEL 1024
#define D_HID   256
#define VOCAB   32000
#define NTOK    4096
#define SEQ     2048
#define LOWR    10
#define LN_EPS  1e-5f

__device__ __forceinline__ float gelu_exact(float v) {
    return 0.5f * v * (1.0f + erff(v * 0.70710678118654752440f));
}

// ---------------------------------------------------------------------------
// small elementwise / reduction kernels (bandwidth-trivial, fp32)
// ---------------------------------------------------------------------------

// dst[n*K+k] = (f16)src[k*N+n]   (weight convert + transpose so B is [N,K])
__global__ void k_convert_transpose(const float* __restrict__ src, f16* __restrict__ dst,
                                    int K, int N) {
    long long idx = (long long)blockIdx.x * blockDim.x + threadIdx.x;
    long long total = (long long)K * N;
    if (idx >= total) return;
    int n = (int)(idx / K);
    int k = (int)(idx % K);
    dst[idx] = (f16)src[(long long)k * N + n];
}

__global__ void k_f32_to_f16(const float* __restrict__ src, f16* __restrict__ dst, int n) {
    int i = blockIdx.x * blockDim.x + threadIdx.x;
    if (i < n) dst[i] = (f16)src[i];
}

// h[tok,:] = embed1[id,:4] @ embed2[4,D]
__global__ void k_embed(const int* __restrict__ tok, const float* __restrict__ e1,
                        const float* __restrict__ e2, float* __restrict__ h) {
    int tIdx = blockIdx.x;
    int id = tok[tIdx];
    const float* row = e1 + (long long)id * 4;
    float r0 = row[0], r1 = row[1], r2 = row[2], r3 = row[3];
    for (int d = threadIdx.x; d < D_MODEL; d += blockDim.x) {
        h[(long long)tIdx * D_MODEL + d] =
            r0 * e2[d] + r1 * e2[D_MODEL + d] + r2 * e2[2 * D_MODEL + d] + r3 * e2[3 * D_MODEL + d];
    }
}

__device__ __forceinline__ float blk_sum(float v, float* red, int tid) {
    __syncthreads();            // protect previous broadcast read of red[0]
    red[tid] = v;
    __syncthreads();
#pragma unroll
    for (int o = 128; o > 0; o >>= 1) {
        if (tid < o) red[tid] += red[tid + o];
        __syncthreads();
    }
    return red[0];
}

// LayerNorm -> hn (fp32) and t[tok,r] = sum_d hn[d]*V[d,r]  (r<10)
__global__ __launch_bounds__(256)
void k_ln1(const float* __restrict__ x, const float* __restrict__ g, const float* __restrict__ b,
           const float* __restrict__ V, float* __restrict__ hn, float* __restrict__ t) {
    __shared__ float red[256];
    int tokI = blockIdx.x, tid = threadIdx.x;
    const float* xr = x + (long long)tokI * D_MODEL;
    float4 vv = ((const float4*)xr)[tid];
    float s  = vv.x + vv.y + vv.z + vv.w;
    float s2 = vv.x*vv.x + vv.y*vv.y + vv.z*vv.z + vv.w*vv.w;
    float mu  = blk_sum(s,  red, tid) * (1.0f / D_MODEL);
    float var = blk_sum(s2, red, tid) * (1.0f / D_MODEL) - mu * mu;
    float rstd = rsqrtf(var + LN_EPS);
    int d0 = tid * 4;
    float o[4] = { (vv.x-mu)*rstd*g[d0+0]+b[d0+0], (vv.y-mu)*rstd*g[d0+1]+b[d0+1],
                   (vv.z-mu)*rstd*g[d0+2]+b[d0+2], (vv.w-mu)*rstd*g[d0+3]+b[d0+3] };
    float* hr = hn + (long long)tokI * D_MODEL;
    ((float4*)hr)[tid] = make_float4(o[0], o[1], o[2], o[3]);
#pragma unroll
    for (int r = 0; r < LOWR; ++r) {
        float p = o[0]*V[(d0+0)*LOWR + r] + o[1]*V[(d0+1)*LOWR + r]
                + o[2]*V[(d0+2)*LOWR + r] + o[3]*V[(d0+3)*LOWR + r];
        float tr = blk_sum(p, red, tid);
        if (tid == 0) t[tokI * 16 + r] = tr;
    }
}

// LayerNorm -> f16 output (GEMM A operand)
__global__ __launch_bounds__(256)
void k_ln2(const float* __restrict__ x, const float* __restrict__ g, const float* __restrict__ b,
           f16* __restrict__ out) {
    __shared__ float red[256];
    int tokI = blockIdx.x, tid = threadIdx.x;
    const float* xr = x + (long long)tokI * D_MODEL;
    float4 vv = ((const float4*)xr)[tid];
    float s  = vv.x + vv.y + vv.z + vv.w;
    float s2 = vv.x*vv.x + vv.y*vv.y + vv.z*vv.z + vv.w*vv.w;
    float mu  = blk_sum(s,  red, tid) * (1.0f / D_MODEL);
    float var = blk_sum(s2, red, tid) * (1.0f / D_MODEL) - mu * mu;
    float rstd = rsqrtf(var + LN_EPS);
    int d0 = tid * 4;
    f16* orow = out + (long long)tokI * D_MODEL;
    orow[d0+0] = (f16)((vv.x-mu)*rstd*g[d0+0]+b[d0+0]);
    orow[d0+1] = (f16)((vv.y-mu)*rstd*g[d0+1]+b[d0+1]);
    orow[d0+2] = (f16)((vv.z-mu)*rstd*g[d0+2]+b[d0+2]);
    orow[d0+3] = (f16)((vv.w-mu)*rstd*g[d0+3]+b[d0+3]);
}

// x += diag*hn + sub*hn[s-1] + super*hn[s+1] + U @ t
__global__ __launch_bounds__(256)
void k_tri_lr(float* __restrict__ x, const float* __restrict__ hn,
              const float* __restrict__ dg, const float* __restrict__ sup,
              const float* __restrict__ sub, const float* __restrict__ U,
              const float* __restrict__ t) {
    int tokI = blockIdx.x, tid = threadIdx.x;
    int s = tokI & (SEQ - 1);
    __shared__ float ts[LOWR];
    if (tid < LOWR) ts[tid] = t[tokI * 16 + tid];
    __syncthreads();
    const float* hr = hn + (long long)tokI * D_MODEL;
    const float* hp = hr - D_MODEL;
    const float* hx = hr + D_MODEL;
    bool hasP = (s > 0), hasN = (s < SEQ - 1);
    for (int d = tid; d < D_MODEL; d += 256) {
        float acc = dg[d] * hr[d];
        if (hasP) acc += sub[d] * hp[d];
        if (hasN) acc += sup[d] * hx[d];
        float lr = 0.f;
#pragma unroll
        for (int r = 0; r < LOWR; ++r) lr += ts[r] * U[d * LOWR + r];
        x[(long long)tokI * D_MODEL + d] += acc + lr;
    }
}

// ---------------------------------------------------------------------------
// WMMA f16 GEMM:  out[M,N] = A[M,K] @ Bt[N,K]^T  (+bias, opt GELU, opt resid)
// 256 threads = 8 wave32; 128x128 block tile; BK=32; wave tile 32x64 (2x4 WMMA)
// ---------------------------------------------------------------------------
template <bool GELU, bool RES, bool OUTF16>
__global__ __launch_bounds__(256)
void k_gemm_wmma(const f16* __restrict__ A, const f16* __restrict__ Bt,
                 const float* __restrict__ bias, const float* __restrict__ resid,
                 float* __restrict__ outF, f16* __restrict__ outH,
                 int N, int K) {
    __shared__ f16 As[128][40];   // +8 pad keeps 16B alignment, staggers banks
    __shared__ f16 Bs[128][40];
    int tid  = threadIdx.x;
    int lane = tid & 31, wid = tid >> 5;
    int lh = lane >> 4, lm = lane & 15;           // lane-half, lane-mod16
    int wm = wid >> 1,  wn = wid & 1;             // 4x2 wave grid
    int m0 = blockIdx.y * 128;
    int n0 = blockIdx.x * 128;

    v8f acc[2][4];
#pragma unroll
    for (int i = 0; i < 2; ++i)
#pragma unroll
        for (int j = 0; j < 4; ++j) acc[i][j] = {};

    for (int kb = 0; kb < K; kb += 32) {
        // stage 128x32 f16 tiles of A and Bt (both row-major along K)
#pragma unroll
        for (int p = 0; p < 2; ++p) {
            int idx = (p * 256 + tid) * 8;        // 8 halfs (16B) per thread per pass
            int r = idx >> 5, c = idx & 31;
            *(uint4*)&As[r][c] = *(const uint4*)(A  + (long long)(m0 + r) * K + kb + c);
            *(uint4*)&Bs[r][c] = *(const uint4*)(Bt + (long long)(n0 + r) * K + kb + c);
        }
        __syncthreads();

        // per-lane fragments per ISA 7.12.2:
        //   A (16x32, 16-bit): elem e -> K = (e/8)*16 + lh*8 + e%8, M = lm
        //   B (32x16, 16-bit): elem e -> K = (e/8)*16 + lh*8 + e%8, N = lm
        union { v16h v; uint4 q[2]; } af[2], bf[4];
#pragma unroll
        for (int tm = 0; tm < 2; ++tm) {
            int row = wm * 32 + tm * 16 + lm;
            af[tm].q[0] = *(const uint4*)&As[row][lh * 8];
            af[tm].q[1] = *(const uint4*)&As[row][16 + lh * 8];
        }
#pragma unroll
        for (int tn = 0; tn < 4; ++tn) {
            int row = wn * 64 + tn * 16 + lm;
            bf[tn].q[0] = *(const uint4*)&Bs[row][lh * 8];
            bf[tn].q[1] = *(const uint4*)&Bs[row][16 + lh * 8];
        }
#pragma unroll
        for (int tm = 0; tm < 2; ++tm)
#pragma unroll
            for (int tn = 0; tn < 4; ++tn)
                acc[tm][tn] = __builtin_amdgcn_wmma_f32_16x16x32_f16(
                    false, af[tm].v, false, bf[tn].v, (short)0, acc[tm][tn], false, false);
        __syncthreads();
    }

    // epilogue: C/D layout -> row = g + lh*8 (within 16x16), col = lm
#pragma unroll
    for (int tm = 0; tm < 2; ++tm)
#pragma unroll
        for (int tn = 0; tn < 4; ++tn)
#pragma unroll
            for (int g = 0; g < 8; ++g) {
                int row = m0 + wm * 32 + tm * 16 + lh * 8 + g;
                int col = n0 + wn * 64 + tn * 16 + lm;
                float v = acc[tm][tn][g] + bias[col];
                if (GELU) v = gelu_exact(v);
                if (RES)  v += resid[(long long)row * N + col];
                if (OUTF16) outH[(long long)row * N + col] = (f16)v;
                else        outF[(long long)row * N + col] = v;
            }
}

// ---------------------------------------------------------------------------
// launch
// ---------------------------------------------------------------------------
extern "C" void kernel_launch(void* const* d_in, const int* in_sizes, int n_in,
                              void* d_out, int out_size, void* d_ws, size_t ws_size,
                              hipStream_t stream) {
    // input order: x, embed1, embed2, 4x{diag,super,sub,U,V,ln1_g,ln1_b,ln2_g,ln2_b,
    //              ffn_w1,ffn_b1,ffn_w2,ffn_b2}, head_w1, head_b1, head_w2, head_b2
    const int*   x  = (const int*)  d_in[0];
    const float* e1 = (const float*)d_in[1];
    const float* e2 = (const float*)d_in[2];
    const float* head_w1 = (const float*)d_in[55];
    const float* head_b1 = (const float*)d_in[56];
    const float* head_w2 = (const float*)d_in[57];
    const float* head_b2 = (const float*)d_in[58];

    // workspace layout (bytes)
    char* W = (char*)d_ws;
    float* h   = (float*)(W + 0);                 // [4096,1024] f32      16 MB
    float* hn  = (float*)(W + 16777216);          // [4096,1024] f32      16 MB
    f16*   h2n = (f16*)  (W + 33554432);          // [4096,1024] f16       8 MB
    f16*   mid = (f16*)  (W + 41943040);          // [4096, 256] f16       2 MB
    float* tbuf= (float*)(W + 44040192);          // [4096, 16]  f32     256 KB
    f16*   wb  = (f16*)  (W + 44302336);          // transposed f16 weights ~21 MB
    const size_t need = 44302336 + 2ull * (4ull * 524288 + 262144 + 8192000);
    if (ws_size < need) return;

    f16* hw1t = wb + 4 * 524288;                  // [256,1024]
    f16* hw2t = hw1t + 262144;                    // [32000,256]

    // 1) convert+transpose all GEMM weights to f16 [N,K]
    for (int L = 0; L < 4; ++L) {
        const float* w1 = (const float*)d_in[3 + 13 * L + 9];
        const float* w2 = (const float*)d_in[3 + 13 * L + 11];
        k_convert_transpose<<<(D_MODEL * D_HID + 255) / 256, 256, 0, stream>>>(
            w1, wb + L * 524288, D_MODEL, D_HID);
        k_convert_transpose<<<(D_HID * D_MODEL + 255) / 256, 256, 0, stream>>>(
            w2, wb + L * 524288 + 262144, D_HID, D_MODEL);
    }
    k_convert_transpose<<<(D_MODEL * D_HID + 255) / 256, 256, 0, stream>>>(
        head_w1, hw1t, D_MODEL, D_HID);
    k_convert_transpose<<<(D_HID * VOCAB + 255) / 256, 256, 0, stream>>>(
        head_w2, hw2t, D_HID, VOCAB);

    // 2) embedding
    k_embed<<<NTOK, 256, 0, stream>>>(x, e1, e2, h);

    dim3 g1(D_HID / 128, NTOK / 128);    // (2, 32)
    dim3 g2(D_MODEL / 128, NTOK / 128);  // (8, 32)
    dim3 gh(VOCAB / 128, NTOK / 128);    // (250, 32)

    // 3) layers
    for (int L = 0; L < 4; ++L) {
        const int base = 3 + 13 * L;
        const float* dg   = (const float*)d_in[base + 0];
        const float* sup  = (const float*)d_in[base + 1];
        const float* sub  = (const float*)d_in[base + 2];
        const float* U    = (const float*)d_in[base + 3];
        const float* V    = (const float*)d_in[base + 4];
        const float* ln1g = (const float*)d_in[base + 5];
        const float* ln1b = (const float*)d_in[base + 6];
        const float* ln2g = (const float*)d_in[base + 7];
        const float* ln2b = (const float*)d_in[base + 8];
        const float* b1   = (const float*)d_in[base + 10];
        const float* b2   = (const float*)d_in[base + 12];
        f16* w1t = wb + L * 524288;
        f16* w2t = w1t + 262144;

        k_ln1<<<NTOK, 256, 0, stream>>>(h, ln1g, ln1b, V, hn, tbuf);
        k_tri_lr<<<NTOK, 256, 0, stream>>>(h, hn, dg, sup, sub, U, tbuf);
        k_ln2<<<NTOK, 256, 0, stream>>>(h, ln2g, ln2b, h2n);
        k_gemm_wmma<true, false, true><<<g1, 256, 0, stream>>>(
            h2n, w1t, b1, nullptr, nullptr, mid, D_HID, D_MODEL);
        k_gemm_wmma<false, true, false><<<g2, 256, 0, stream>>>(
            mid, w2t, b2, h, h, nullptr, D_MODEL, D_HID);
    }

    // 4) head
    k_f32_to_f16<<<(NTOK * D_MODEL + 255) / 256, 256, 0, stream>>>(h, h2n, NTOK * D_MODEL);
    k_gemm_wmma<true, false, true><<<g1, 256, 0, stream>>>(
        h2n, hw1t, head_b1, nullptr, nullptr, mid, D_HID, D_MODEL);
    k_gemm_wmma<false, false, false><<<gh, 256, 0, stream>>>(
        mid, hw2t, head_b2, nullptr, (float*)d_out, nullptr, VOCAB, D_HID);
}
